// MultiheadAttention_68195490726014
// MI455X (gfx1250) — matrix operand order repaired and checked by reading, not compiled, not run on hardware
//
#include <hip/hip_runtime.h>

#define NPOINT 1024
#define NHEADS 8
#define EDIM   64
#define FEAT   256
#define BATCH  8
#define C3     (EDIM * NHEADS * 3)   // 1536

typedef __attribute__((ext_vector_type(16))) __bf16        v16bf;
typedef __attribute__((ext_vector_type(8)))  float         v8f;
typedef __attribute__((ext_vector_type(4)))  unsigned int  v4u;

union Frag16 { v16bf v; unsigned int u[8]; v4u q[2]; };

__device__ __forceinline__ unsigned short f32_to_bf16(float f) {
    unsigned int u = __float_as_uint(f);
    unsigned int r = u + 0x7FFFu + ((u >> 16) & 1u);   // round-to-nearest-even
    return (unsigned short)(r >> 16);
}
__device__ __forceinline__ unsigned int pack_bf16(float lo, float hi) {
#if defined(__has_builtin) && __has_builtin(__builtin_amdgcn_cvt_pk_bf16_f32)
    typedef __attribute__((ext_vector_type(2))) __bf16 v2bf;
    union { v2bf v; unsigned int u; } c;
    c.v = __builtin_amdgcn_cvt_pk_bf16_f32(lo, hi);
    return c.u;
#else
    return (unsigned int)f32_to_bf16(lo) | ((unsigned int)f32_to_bf16(hi) << 16);
#endif
}

// ---------------------------------------------------------------------------
// Kernel 0a: x (8192x256 f32) -> x_bf (row-major bf16), packed dwords.
// ---------------------------------------------------------------------------
__global__ __launch_bounds__(256) void cvt_x(const float* __restrict__ x,
                                             unsigned int* __restrict__ x_bf) {
    const int total = BATCH * NPOINT * FEAT / 2;       // dword count
    for (int i = blockIdx.x * 256 + threadIdx.x; i < total; i += gridDim.x * 256)
        x_bf[i] = pack_bf16(x[2 * i], x[2 * i + 1]);
}

// ---------------------------------------------------------------------------
// Kernel 0b: W (256x1536 f32) -> Wt_bf (transposed: [col][f], bf16 dwords).
// Transposing makes B-fragment K-pairs (f, f+1) contiguous.
// ---------------------------------------------------------------------------
__global__ __launch_bounds__(256) void cvt_Wt(const float* __restrict__ W,
                                              unsigned int* __restrict__ Wt_bf) {
    const int total = C3 * (FEAT / 2);                 // 1536 * 128 dwords
    for (int i = blockIdx.x * 256 + threadIdx.x; i < total; i += gridDim.x * 256) {
        int col = i >> 7;                              // /128
        int fp  = i & 127;                             // f pair index
        Wt_bf[i] = pack_bf16(W[(size_t)(2 * fp) * C3 + col],
                             W[(size_t)(2 * fp + 1) * C3 + col]);
    }
}

// ---------------------------------------------------------------------------
// Kernel 1: emb = x_bf @ W  (all-bf16 operands, f32 accum) -> Q,K [b][h][n][e],
// V^T [b][h][e][n].  Wave owns one (s,h,e-tile) group: all indices uniform,
// hot loop is 4x global_load_b128 + 1 WMMA per 32-wide K chunk.
// ---------------------------------------------------------------------------
__global__ __launch_bounds__(256) void qkv_gemm(
        const unsigned int* __restrict__ x_bf, const unsigned int* __restrict__ Wt_bf,
        unsigned short* __restrict__ Qb, unsigned short* __restrict__ Kb,
        unsigned short* __restrict__ VTb) {
    const int lane = threadIdx.x & 31;
    const int wave = threadIdx.x >> 5;
    const int lh   = lane & 15;
    const int hi   = lane >> 4;
    const int rt   = blockIdx.x;                       // 0..511
    const int ct   = blockIdx.y * 8 + wave;            // 0..95 -> (s,h,et)
    const int et = ct & 3;
    const int h  = (ct >> 2) & 7;
    const int s  = __builtin_amdgcn_readfirstlane(ct >> 5);
    const int e0 = et * 16;
    const int row0 = rt * 16;

    const int wcol = (e0 + lh) * 24 + h * 3 + s;       // emb column for this lane
    const unsigned int* xrow = x_bf + (size_t)(row0 + lh) * (FEAT / 2);
    const unsigned int* wrow = Wt_bf + (size_t)wcol * (FEAT / 2);

    v8f acc = {};
    #pragma unroll
    for (int kc2 = 0; kc2 < FEAT / 2; kc2 += 16) {     // kc2 = kc/2 (dwords)
        Frag16 a, b;
        a.q[0] = *(const v4u*)(xrow + kc2 + 8 * hi);       // A elems 0..3
        a.q[1] = *(const v4u*)(xrow + kc2 + 8 * hi + 4);   // A elems 4..7
        b.q[0] = *(const v4u*)(wrow + kc2 + 8 * hi);       // B elems 0..3
        b.q[1] = *(const v4u*)(wrow + kc2 + 8 * hi + 4);   // B elems 4..7
        acc = __builtin_amdgcn_wmma_f32_16x16x32_bf16(
                  false, a.v, false, b.v, (short)0, acc, false, false);
    }

    const int bb = row0 >> 10;
    const int nbase = row0 & 1023;
    const size_t bhOff = (size_t)(bb * NHEADS + h) * (size_t)(NPOINT * EDIM);

    if (s == 0 || s == 1) {                            // Q/K: [n][e]
        unsigned short* dst = (s == 0 ? Qb : Kb) + bhOff + (e0 + lh);
        #pragma unroll
        for (int r = 0; r < 8; ++r) {
            int n = nbase + r + (hi << 3);
            dst[(size_t)n * EDIM] = f32_to_bf16(acc[r]);
        }
    } else {                                           // V^T: [e][n]
        unsigned short* dst = VTb + bhOff + (size_t)(e0 + lh) * NPOINT + nbase;
        #pragma unroll
        for (int r = 0; r < 8; ++r)
            dst[r + (hi << 3)] = f32_to_bf16(acc[r]);
    }
}

// ---------------------------------------------------------------------------
// Kernel 2: per (b, 16-row tile), 8 waves = 8 heads.
// ---------------------------------------------------------------------------
__global__ __launch_bounds__(256) void attention(
        const unsigned short* __restrict__ Qb,
        const unsigned short* __restrict__ Kb,
        const unsigned short* __restrict__ VTb,
        float* __restrict__ attn_out, float* __restrict__ out_out) {
    extern __shared__ char smem_raw[];
    float*          attnLds = (float*)smem_raw;                            // 64 KB
    unsigned short* pStage  = (unsigned short*)(smem_raw + 16 * 1024 * 4); //  8 KB

    const int lane = threadIdx.x & 31;
    const int wave = threadIdx.x >> 5;          // head
    const int lh   = lane & 15;
    const int hi   = lane >> 4;
    const int bb   = blockIdx.x >> 6;
    const int n0   = (blockIdx.x & 63) * 16;

    for (int i = threadIdx.x; i < 16 * 1024; i += 256) attnLds[i] = 0.0f;
    __syncthreads();

    const size_t bh = (size_t)(bb * NHEADS + wave) * (size_t)(NPOINT * EDIM);
    const unsigned int* Qw  = (const unsigned int*)(Qb  + bh);
    const unsigned int* Kw  = (const unsigned int*)(Kb  + bh);
    const unsigned int* VTw = (const unsigned int*)(VTb + bh);

    // Q block (16 x 64) -> two resident A fragments (4x b128)
    Frag16 qa[2];
    {
        const unsigned int* qrow = Qw + (size_t)(n0 + lh) * (EDIM / 2);
        #pragma unroll
        for (int half = 0; half < 2; ++half) {
            qa[half].q[0] = *(const v4u*)(qrow + half * 16 + 8 * hi);
            qa[half].q[1] = *(const v4u*)(qrow + half * 16 + 8 * hi + 4);
        }
    }

    // ---- Pass 1: per-lane online max / rescaled sum (no in-loop shuffles) ----
    float ml[8], ll[8];
    #pragma unroll
    for (int r = 0; r < 8; ++r) { ml[r] = -3.0e38f; ll[r] = 0.0f; }

    for (int j = 0; j < 64; ++j) {
        const unsigned int* krow = Kw + (size_t)(j * 16 + lh) * (EDIM / 2);
        __builtin_prefetch(krow + ((j < 63) ? 512 : 0), 0, 1);  // next K row tile
        v8f sv = {};
        #pragma unroll
        for (int half = 0; half < 2; ++half) {
            Frag16 kb;
            kb.q[0] = *(const v4u*)(krow + half * 16 + 8 * hi);
            kb.q[1] = *(const v4u*)(krow + half * 16 + 8 * hi + 4);
            sv = __builtin_amdgcn_wmma_f32_16x16x32_bf16(
                     false, qa[half].v, false, kb.v, (short)0, sv, false, false);
        }
        #pragma unroll
        for (int r = 0; r < 8; ++r) {
            float v  = sv[r];
            float mn = fmaxf(ml[r], v);
            ll[r] = ll[r] * __expf(ml[r] - mn) + __expf(v - mn);
            ml[r] = mn;
        }
    }
    float m[8], rsc[8];
    #pragma unroll
    for (int r = 0; r < 8; ++r) {
        float mx = ml[r];
        #pragma unroll
        for (int off = 1; off < 16; off <<= 1)
            mx = fmaxf(mx, __shfl_xor(mx, off, 32));
        float c = ll[r] * __expf(ml[r] - mx);
        #pragma unroll
        for (int off = 1; off < 16; off <<= 1)
            c += __shfl_xor(c, off, 32);
        m[r]   = mx;
        rsc[r] = 0.125f / c;                    // 1/sqrt(E) / rowsum
    }

    // ---- Pass 2: P tiles -> attn accumulation + out = P * V^T ----
    v8f o[4] = {};
    unsigned short* myStage = pStage + wave * (16 * 32);

    for (int t = 0; t < 32; ++t) {
        #pragma unroll
        for (int jj = 0; jj < 2; ++jj) {
            int j = t * 2 + jj;
            const unsigned int* krow = Kw + (size_t)(j * 16 + lh) * (EDIM / 2);
            __builtin_prefetch(krow + ((j < 63) ? 512 : 0), 0, 1);
            v8f sv = {};
            #pragma unroll
            for (int half = 0; half < 2; ++half) {
                Frag16 kb;
                kb.q[0] = *(const v4u*)(krow + half * 16 + 8 * hi);
                kb.q[1] = *(const v4u*)(krow + half * 16 + 8 * hi + 4);
                sv = __builtin_amdgcn_wmma_f32_16x16x32_bf16(
                         false, qa[half].v, false, kb.v, (short)0, sv, false, false);
            }
            #pragma unroll
            for (int r = 0; r < 8; ++r) {
                int   M = r + (hi << 3);
                float p = __expf(sv[r] - m[r]) * rsc[r];
                atomicAdd(&attnLds[M * 1024 + j * 16 + lh], p); // head sum (ds_add_f32)
                myStage[M * 32 + jj * 16 + lh] = f32_to_bf16(p);
            }
        }
        // P (16x32) back from wave-private LDS in A-layout (2x ds_load_b128)
        Frag16 pa;
        {
            const unsigned int* prow = (const unsigned int*)(myStage + (size_t)lh * 32);
            pa.q[0] = *(const v4u*)(prow + 8 * hi);
            pa.q[1] = *(const v4u*)(prow + 8 * hi + 4);
        }
        // out(16x64) += P(16x32) * V^T-slice(32x64)
        #pragma unroll
        for (int et = 0; et < 4; ++et) {
            Frag16 vb;
            const unsigned int* vrow = VTw + (size_t)(et * 16 + lh) * (NPOINT / 2);
            vb.q[0] = *(const v4u*)(vrow + t * 16 + 8 * hi);
            vb.q[1] = *(const v4u*)(vrow + t * 16 + 8 * hi + 4);
            o[et] = __builtin_amdgcn_wmma_f32_16x16x32_bf16(
                        false, pa.v, false, vb.v, (short)0, o[et], false, false);
        }
    }

    // out writeout: (B, N, H*E) fp32
    #pragma unroll
    for (int et = 0; et < 4; ++et)
        #pragma unroll
        for (int r = 0; r < 8; ++r) {
            int M = r + (hi << 3);
            size_t idx = ((size_t)bb * NPOINT + (n0 + M)) * (size_t)(NHEADS * EDIM)
                       + (size_t)wave * EDIM + et * 16 + lh;
            out_out[idx] = o[et][r];
        }

    __syncthreads();
    const float invH = 1.0f / (float)NHEADS;
    for (int i = threadIdx.x; i < 16 * 1024; i += 256) {
        int M = i >> 10, c = i & 1023;
        attn_out[((size_t)bb * NPOINT + (n0 + M)) * NPOINT + c] = attnLds[i] * invH;
    }
}

// ---------------------------------------------------------------------------
extern "C" void kernel_launch(void* const* d_in, const int* in_sizes, int n_in,
                              void* d_out, int out_size, void* d_ws, size_t ws_size,
                              hipStream_t stream) {
    (void)in_sizes; (void)n_in; (void)out_size; (void)ws_size;
    const float* x = (const float*)d_in[0];
    const float* W = (const float*)d_in[1];

    float* attn_out = (float*)d_out;
    float* out_out  = (float*)d_out + (size_t)BATCH * NPOINT * NPOINT;

    const size_t qkvElems = (size_t)BATCH * NHEADS * NPOINT * EDIM;  // 4M each
    unsigned short* Qb   = (unsigned short*)d_ws;
    unsigned short* Kb   = Qb + qkvElems;
    unsigned short* VTb  = Kb + qkvElems;
    unsigned int*   x_bf = (unsigned int*)(VTb + qkvElems);              // 4 MB
    unsigned int*   Wt_bf= x_bf + (size_t)BATCH * NPOINT * FEAT / 2;     // 768 KB

    hipLaunchKernelGGL(cvt_x,  dim3(512), dim3(256), 0, stream, x, x_bf);
    hipLaunchKernelGGL(cvt_Wt, dim3(256), dim3(256), 0, stream, W, Wt_bf);

    hipLaunchKernelGGL(qkv_gemm, dim3(512, 12), dim3(256), 0, stream,
                       x_bf, Wt_bf, Qb, Kb, VTb);

    const size_t smem = 16 * 1024 * sizeof(float)
                      + 8 * 16 * 32 * sizeof(unsigned short);
    hipFuncSetAttribute((const void*)attention,
                        hipFuncAttributeMaxDynamicSharedMemorySize, (int)smem);
    hipLaunchKernelGGL(attention, dim3(BATCH * (NPOINT / 16)), dim3(256), smem, stream,
                       Qb, Kb, VTb, attn_out, out_out);
}
